// YOLOv1_33191507263940
// MI455X (gfx1250) — matrix operand order repaired and checked by reading, not compiled, not run on hardware
//
#include <hip/hip_runtime.h>

// ---------- vector types ----------
typedef __bf16          bf16x16 __attribute__((ext_vector_type(16)));
typedef float           f32x8   __attribute__((ext_vector_type(8)));
typedef float           f32x4   __attribute__((ext_vector_type(4)));
typedef unsigned short  u16x16  __attribute__((ext_vector_type(16)));
typedef unsigned short  u16x8   __attribute__((ext_vector_type(8)));
typedef int             si32x4  __attribute__((ext_vector_type(4)));
typedef int             si32x8  __attribute__((ext_vector_type(8)));

union Frag16 { u16x16 u; bf16x16 b; };

// ---------- bf16 helpers (raw-bit storage; RNE convert) ----------
__device__ __forceinline__ unsigned short f2bf(float f) {
    unsigned int u = __float_as_uint(f);
    u += 0x7FFFu + ((u >> 16) & 1u);          // round-to-nearest-even
    return (unsigned short)(u >> 16);
}
__device__ __forceinline__ float bf2f(unsigned short h) {
    return __uint_as_float(((unsigned int)h) << 16);
}
__device__ __forceinline__ float ldval(const float* p, int i)          { return p[i]; }
__device__ __forceinline__ float ldval(const unsigned short* p, int i) { return bf2f(p[i]); }

// ---------------------------------------------------------------------------
// Tensor Data Mover: async DMA of a 2-D fp32 tile (tileD1 rows x tileD0 cols,
// row stride strideElems) from global into LDS.  D# per CDNA5 ISA:
//   group0: [1:0]=count=1, [63:32]=lds_addr, [120:64]=global_addr, [127:126]=2
//   group1: data_size=2(4B), tensor_dim0/1 (OOB rows/cols zero-fill),
//           tile_dim0/1, tensor_dim0_stride.
// Issued via inline asm (portable across toolchains; builtin arity differs).
// Tracked on TENSORcnt -> s_wait_tensorcnt 0 before consuming.
// ---------------------------------------------------------------------------
__device__ __forceinline__ void tdm_load_2d_f32(void* ldsDst, const void* gSrc,
                                                int tensorD0, int tensorD1,
                                                int tileD0, int tileD1,
                                                long strideElems)
{
    const unsigned long long ga = (unsigned long long)(uintptr_t)gSrc;
    const unsigned int lds = (unsigned int)(uintptr_t)ldsDst;   // LDS byte offset = addr[31:0]
    si32x4 g0;
    g0[0] = 1;                                                  // count=1 (valid user D#)
    g0[1] = (int)lds;                                           // lds_addr
    g0[2] = (int)(unsigned int)(ga & 0xFFFFFFFFu);              // global_addr[31:0]
    g0[3] = (int)((((unsigned int)(ga >> 32)) & 0x01FFFFFFu) | 0x80000000u); // addr[56:32] | type=2
    const unsigned long long st = (unsigned long long)strideElems;
    si32x8 g1;
    g1[0] = (2 << 16);                                          // data_size = 4 bytes
    g1[1] = (int)(((unsigned)tensorD0 & 0xFFFFu) << 16);        // tensor_dim0[15:0]
    g1[2] = (int)((((unsigned)tensorD0 >> 16) & 0xFFFFu) |
                  (((unsigned)tensorD1 & 0xFFFFu) << 16));      // dim0[31:16] | dim1[15:0]
    g1[3] = (int)((((unsigned)tensorD1 >> 16) & 0xFFFFu) |
                  (((unsigned)tileD0) << 16));                  // dim1[31:16] | tile_dim0
    g1[4] = (int)((unsigned)tileD1 & 0xFFFFu);                  // tile_dim1 (tile_dim2=0)
    g1[5] = (int)(unsigned int)(st & 0xFFFFFFFFu);              // dim0_stride[31:0]
    g1[6] = (int)((unsigned int)(st >> 32) & 0xFFFFu);          // dim0_stride[47:32]
    g1[7] = 0;
    asm volatile("tensor_load_to_lds %0, %1" :: "s"(g0), "s"(g1) : "memory");
}

// ---------------------------------------------------------------------------
// Implicit-GEMM conv, D = W(16xK) * im2col(Kx16), two v_wmma_f32_16x16x32_bf16
// per staged K=64 panel.  The 16x64 fp32 weight panel is DMA'd into LDS by the
// Tensor Data Mover (wave 0 issues; tensor_dim0 = remaining K so the hardware
// zero-fills the K tail), all 8 waves read their A fragments with ds_read_b128
// and convert to bf16; each wave gathers its own 16-pixel B fragment.
// ---------------------------------------------------------------------------
template <typename TIn>
__global__ __launch_bounds__(256)
void conv_wmma(const TIn* __restrict__ in, const float* __restrict__ w,
               const float* __restrict__ bias, unsigned short* __restrict__ out,
               int N, int Cin, int H, int W, int Cout,
               int KH, int KW, int stride, int pad, int OH, int OW)
{
    __shared__ __align__(16) float Asmem[16 * 64];            // 16 Cout rows x 64 K (fp32)

    const int lane    = threadIdx.x & 31;
    const int wv      = threadIdx.x >> 5;
    const int tileCol = blockIdx.x * 8 + wv;                  // 16-pixel tile
    const int coBase  = blockIdx.y * 16;                      // Cout tile
    const int KHW     = KH * KW;
    const int Ktot    = Cin * KHW;
    const int Ptot    = N * OH * OW;

    const int colq    = lane & 15;
    const int hiHalf  = lane >> 4;                            // 0 or 1

    const int p       = tileCol * 16 + colq;
    const bool pvalid = (p < Ptot);
    int n = 0, oh = 0, ow = 0;
    if (pvalid) { n = p / (OH * OW); int rp = p % (OH * OW); oh = rp / OW; ow = rp % OW; }
    const int ohs = oh * stride - pad;
    const int ows = ow * stride - pad;

    f32x8 acc = {};
    const int ksteps = (Ktot + 63) / 64;

    for (int s = 0; s < ksteps; ++s) {
        const int k0 = s * 64;
        __syncthreads();                                      // protect previous panel reads
        if (wv == 0) {
            tdm_load_2d_f32(&Asmem[0], w + coBase * Ktot + k0,
                            /*tensorD0=*/Ktot - k0, /*tensorD1=*/16,
                            /*tileD0=*/64, /*tileD1=*/16, /*stride=*/Ktot);
            __builtin_amdgcn_s_wait_tensorcnt(0);
        }
        __syncthreads();

        #pragma unroll
        for (int t = 0; t < 2; ++t) {
            // A fragment: row m=lane&15, 16 contiguous K per lane
            Frag16 a;
            {
                const f32x4* row = (const f32x4*)(&Asmem[(lane & 15) * 64]);
                const int c = t * 8 + hiHalf * 4;
                const f32x4 a0 = row[c];                      // K = base..base+3
                const f32x4 a1 = row[c + 1];                  // K = base+4..base+7
                const f32x4 a2 = row[c + 2];                  // K = base+8..base+11
                const f32x4 a3 = row[c + 3];                  // K = base+12..base+15
                #pragma unroll
                for (int h = 0; h < 4; ++h) {
                    a.u[h]      = f2bf(a0[h]);
                    a.u[h + 4]  = f2bf(a1[h]);
                    a.u[h + 8]  = f2bf(a2[h]);
                    a.u[h + 12] = f2bf(a3[h]);
                }
            }
            // B fragment: im2col gather, 16 contiguous K per lane
            Frag16 bf;
            {
                const int kb = k0 + t * 32 + hiHalf * 16;
                int ci = kb / KHW; int rem = kb - ci * KHW;
                int kh = rem / KW; int kw = rem - kh * KW;
                #pragma unroll
                for (int h = 0; h < 16; ++h) {
                    float v = 0.0f;
                    const int k = kb + h;
                    if (pvalid && k < Ktot) {
                        const int ih = ohs + kh, iw = ows + kw;
                        if ((unsigned)ih < (unsigned)H && (unsigned)iw < (unsigned)W)
                            v = ldval(in, ((n * Cin + ci) * H + ih) * W + iw);
                    }
                    bf.u[h] = f2bf(v);
                    if (++kw == KW) { kw = 0; if (++kh == KH) { kh = 0; ++ci; } }
                }
            }
            acc = __builtin_amdgcn_wmma_f32_16x16x32_bf16(false, a.b, false, bf.b,
                                                          (short)0, acc, false, false);
        }
    }

    if (pvalid) {
        #pragma unroll
        for (int r = 0; r < 8; ++r) {
            const int co = coBase + r + hiHalf * 8;
            const float v = acc[r] + bias[co];
            out[((n * Cout + co) * OH + oh) * OW + ow] = f2bf(v);
        }
    }
}

// ---------------------------------------------------------------------------
// MaxPool (VALID), bf16 in/out.
// ---------------------------------------------------------------------------
__global__ __launch_bounds__(256)
void maxpool_bf16(const unsigned short* __restrict__ in, unsigned short* __restrict__ out,
                  int N, int C, int H, int W, int k, int s, int OH, int OW)
{
    const int idx = blockIdx.x * blockDim.x + threadIdx.x;
    const int total = N * C * OH * OW;
    if (idx >= total) return;
    const int ow = idx % OW; int t = idx / OW;
    const int oh = t % OH;   t /= OH;
    const int c  = t % C;
    const int n  = t / C;
    const int h0 = oh * s, w0 = ow * s;
    float m = -3.402823466e38f;
    for (int i = 0; i < k; ++i)
        for (int j = 0; j < k; ++j)
            m = fmaxf(m, bf2f(in[((n * C + c) * H + h0 + i) * W + w0 + j]));
    out[idx] = f2bf(m);
}

// ---------------------------------------------------------------------------
// Bilinear resize (half-pixel), bf16 in/out.
// ---------------------------------------------------------------------------
__global__ __launch_bounds__(256)
void bilinear_bf16(const unsigned short* __restrict__ in, unsigned short* __restrict__ out,
                   int NC, int IH, int IW, int OH, int OW)
{
    const int idx = blockIdx.x * blockDim.x + threadIdx.x;
    const int total = NC * OH * OW;
    if (idx >= total) return;
    const int ow = idx % OW; int t = idx / OW;
    const int oh = t % OH;
    const int nc = t / OH;

    float cy = (oh + 0.5f) * ((float)IH / (float)OH) - 0.5f;
    cy = fminf(fmaxf(cy, 0.0f), (float)(IH - 1));
    int r0 = (int)floorf(cy); int r1 = r0 + 1; if (r1 > IH - 1) r1 = IH - 1;
    const float wr = cy - (float)r0;

    float cx = (ow + 0.5f) * ((float)IW / (float)OW) - 0.5f;
    cx = fminf(fmaxf(cx, 0.0f), (float)(IW - 1));
    int c0 = (int)floorf(cx); int c1 = c0 + 1; if (c1 > IW - 1) c1 = IW - 1;
    const float wc = cx - (float)c0;

    const unsigned short* base = in + nc * IH * IW;
    const float v00 = bf2f(base[r0 * IW + c0]);
    const float v01 = bf2f(base[r0 * IW + c1]);
    const float v10 = bf2f(base[r1 * IW + c0]);
    const float v11 = bf2f(base[r1 * IW + c1]);
    const float top = v00 * (1.0f - wr) + v10 * wr;
    const float bot = v01 * (1.0f - wr) + v11 * wr;
    out[idx] = f2bf(top * (1.0f - wc) + bot * wc);
}

// ---------------------------------------------------------------------------
// Locally-connected layer: per (h,w) a 16(batch) x 256(out) x 9216 WMMA GEMM.
// Leaky ReLU fused. grid = (49, 16), one wave per block.
// ---------------------------------------------------------------------------
__global__ __launch_bounds__(32)
void localconv_wmma(const unsigned short* __restrict__ x7,    // [16,1024,7,7] bf16
                    const float* __restrict__ lcw,            // [1024,256,7,7,3,3]
                    const float* __restrict__ lcb,            // [256,7,7]
                    unsigned short* __restrict__ out)         // [16,256,7,7] bf16
{
    const int lane   = threadIdx.x & 31;
    const int hiHalf = lane >> 4;
    const int colq   = lane & 15;
    const int hw     = blockIdx.x;            // 0..48
    const int h      = hw / 7, w = hw % 7;
    const int o      = blockIdx.y * 16 + colq;

    f32x8 acc = {};
    for (int k0 = 0; k0 < 9216; k0 += 32) {   // K = 1024*9
        Frag16 a, bf;
        #pragma unroll
        for (int hh = 0; hh < 16; ++hh) {
            const int k   = k0 + hiHalf * 16 + hh;
            const int c   = k / 9, rem = k % 9;
            const int ii  = rem / 3, jj = rem % 3;
            const int ih  = h + ii - 1, iw = w + jj - 1;
            float v = 0.0f;
            if ((unsigned)ih < 7u && (unsigned)iw < 7u)
                v = bf2f(x7[((colq * 1024 + c) * 7 + ih) * 7 + iw]);
            a.u[hh] = f2bf(v);
        }
        #pragma unroll
        for (int hh = 0; hh < 16; ++hh) {
            const int k   = k0 + hiHalf * 16 + hh;
            const int c   = k / 9, rem = k % 9;
            bf.u[hh] = f2bf(lcw[((c * 256 + o) * 49 + hw) * 9 + rem]);
        }
        acc = __builtin_amdgcn_wmma_f32_16x16x32_bf16(false, a.b, false, bf.b,
                                                      (short)0, acc, false, false);
    }

    const float bias = lcb[o * 49 + hw];
    #pragma unroll
    for (int r = 0; r < 8; ++r) {
        const int b = r + hiHalf * 8;
        float v = acc[r] + bias;
        v = v > 0.0f ? v : 0.1f * v;          // leaky ReLU
        out[(b * 256 + o) * 49 + hw] = f2bf(v);
    }
}

// ---------------------------------------------------------------------------
// Final FC: y(16 x 4949) = act(16 x 12544) @ fc_w^T + b.
// Each wave owns one 16-column tile; its 16x64 fp32 fc_w panel is streamed by
// the TDM into a private 4KB LDS slice (tensor_dim1 = 4949-n0 zero-fills the
// ragged tail), A fragments are 16-byte bf16 vector loads straight from global.
// ---------------------------------------------------------------------------
__global__ __launch_bounds__(256)
void fc_wmma(const unsigned short* __restrict__ act,   // [16,12544] bf16
             const float* __restrict__ fcw,            // [4949,12544]
             const float* __restrict__ fcb,            // [4949]
             float* __restrict__ out)                  // [16,4949]
{
    __shared__ __align__(16) float Bsmem[8 * 16 * 64];        // 4KB per wave

    const int lane   = threadIdx.x & 31;
    const int wv     = threadIdx.x >> 5;
    const int tile   = blockIdx.x * 8 + wv;
    if (tile >= 310) return;                  // ceil(4949/16); no barriers below
    const int hiHalf = lane >> 4;
    const int colq   = lane & 15;
    const int n0     = tile * 16;
    const int n      = n0 + colq;
    float* myB = &Bsmem[wv * 16 * 64];
    const unsigned short* arow = act + colq * 12544;

    f32x8 acc = {};
    for (int k0 = 0; k0 < 12544; k0 += 64) {
        tdm_load_2d_f32(myB, fcw + (size_t)n0 * 12544 + k0,
                        /*tensorD0=*/12544 - k0, /*tensorD1=*/4949 - n0,
                        /*tileD0=*/64, /*tileD1=*/16, /*stride=*/12544);
        __builtin_amdgcn_s_wait_tensorcnt(0);

        #pragma unroll
        for (int t = 0; t < 2; ++t) {
            // A: bf16 bits already in fragment-run layout -> two 16B loads
            Frag16 a;
            {
                const u16x8* pa = (const u16x8*)(arow + k0 + t * 32 + hiHalf * 16);
                const u16x8 lo = pa[0];                       // K = base..base+7
                const u16x8 hi = pa[1];                       // K = base+8..base+15
                #pragma unroll
                for (int h = 0; h < 8; ++h) { a.u[h] = lo[h]; a.u[h + 8] = hi[h]; }
            }
            // B: row = column index n, 16 contiguous K fp32 from LDS panel
            Frag16 bf;
            {
                const f32x4* brow = (const f32x4*)(myB + colq * 64);
                const int c = t * 8 + hiHalf * 4;
                const f32x4 b0 = brow[c], b1 = brow[c + 1], b2 = brow[c + 2], b3 = brow[c + 3];
                #pragma unroll
                for (int e = 0; e < 4; ++e) {
                    bf.u[e]      = f2bf(b0[e]);
                    bf.u[e + 4]  = f2bf(b1[e]);
                    bf.u[e + 8]  = f2bf(b2[e]);
                    bf.u[e + 12] = f2bf(b3[e]);
                }
            }
            acc = __builtin_amdgcn_wmma_f32_16x16x32_bf16(false, a.b, false, bf.b,
                                                          (short)0, acc, false, false);
        }
    }
    if (n < 4949) {
        const float bias = fcb[n];
        #pragma unroll
        for (int r = 0; r < 8; ++r) {
            const int b = r + hiHalf * 8;
            out[b * 4949 + n] = acc[r] + bias;
        }
    }
}

// ---------------------------------------------------------------------------
// Orchestration
// ---------------------------------------------------------------------------
static inline int cdiv_i(int a, int b) { return (a + b - 1) / b; }
static inline dim3 conv_grid(int N, int OH, int OW, int Cout) {
    const int tiles = cdiv_i(N * OH * OW, 16);
    return dim3((unsigned)cdiv_i(tiles, 8), (unsigned)(Cout / 16));
}

extern "C" void kernel_launch(void* const* d_in, const int* in_sizes, int n_in,
                              void* d_out, int out_size, void* d_ws, size_t ws_size,
                              hipStream_t stream) {
    (void)in_sizes; (void)n_in; (void)out_size; (void)ws_size;
    const float* x   = (const float*)d_in[0];
    const float* w1  = (const float*)d_in[1];
    const float* b1  = (const float*)d_in[2];
    const float* w3  = (const float*)d_in[3];
    const float* b3  = (const float*)d_in[4];
    const float* w4  = (const float*)d_in[5];
    const float* b4  = (const float*)d_in[6];
    const float* w7  = (const float*)d_in[7];
    const float* b7  = (const float*)d_in[8];
    const float* w8  = (const float*)d_in[9];
    const float* b8  = (const float*)d_in[10];
    const float* wu  = (const float*)d_in[11];
    const float* bu  = (const float*)d_in[12];
    const float* lcw = (const float*)d_in[13];
    const float* lcb = (const float*)d_in[14];
    const float* fcw = (const float*)d_in[15];
    const float* fcb = (const float*)d_in[16];
    float* out = (float*)d_out;

    // ping/pong bf16 activation buffers in workspace
    unsigned short* bufA = (unsigned short*)d_ws;                                      // <= 103 MB
    unsigned short* bufB = (unsigned short*)((char*)d_ws + (size_t)104 * 1024 * 1024); // <= 77 MB

    // conv1 3->64, 7x7 s2 p3 : 448 -> 224
    conv_wmma<float><<<conv_grid(16,224,224,64), 256, 0, stream>>>(
        x, w1, b1, bufA, 16,3,448,448, 64, 7,7, 2,3, 224,224);
    maxpool_bf16<<<cdiv_i(16*64*112*112,256), 256, 0, stream>>>(
        bufA, bufB, 16,64,224,224, 2,2, 112,112);
    // conv3 64->192, 3x3 p1 : 112
    conv_wmma<unsigned short><<<conv_grid(16,112,112,192), 256, 0, stream>>>(
        bufB, w3, b3, bufA, 16,64,112,112, 192, 3,3, 1,1, 112,112);
    maxpool_bf16<<<cdiv_i(16*192*56*56,256), 256, 0, stream>>>(
        bufA, bufB, 16,192,112,112, 2,2, 56,56);
    // conv4 192->256, 3x3 p0 : 56 -> 54
    conv_wmma<unsigned short><<<conv_grid(16,54,54,256), 256, 0, stream>>>(
        bufB, w4, b4, bufA, 16,192,56,56, 256, 3,3, 1,0, 54,54);
    maxpool_bf16<<<cdiv_i(16*256*27*27,256), 256, 0, stream>>>(
        bufA, bufB, 16,256,54,54, 2,2, 27,27);
    // conv7 256->512, 2x2 : 27 -> 26
    conv_wmma<unsigned short><<<conv_grid(16,26,26,512), 256, 0, stream>>>(
        bufB, w7, b7, bufA, 16,256,27,27, 512, 2,2, 1,0, 26,26);
    maxpool_bf16<<<cdiv_i(16*512*13*13,256), 256, 0, stream>>>(
        bufA, bufB, 16,512,26,26, 2,2, 13,13);
    // conv8 512->1024, 2x2 : 13 -> 12
    conv_wmma<unsigned short><<<conv_grid(16,12,12,1024), 256, 0, stream>>>(
        bufB, w8, b8, bufA, 16,512,13,13, 1024, 2,2, 1,0, 12,12);
    // convu x2 1024->1024, 2x2 : 12 -> 11 -> 10
    conv_wmma<unsigned short><<<conv_grid(16,11,11,1024), 256, 0, stream>>>(
        bufA, wu, bu, bufB, 16,1024,12,12, 1024, 2,2, 1,0, 11,11);
    conv_wmma<unsigned short><<<conv_grid(16,10,10,1024), 256, 0, stream>>>(
        bufB, wu, bu, bufA, 16,1024,11,11, 1024, 2,2, 1,0, 10,10);
    // maxpool 5,4 : 10 -> 2   (slice [:14,:14] is a no-op at 2x2)
    maxpool_bf16<<<cdiv_i(16*1024*2*2,256), 256, 0, stream>>>(
        bufA, bufB, 16,1024,10,10, 5,4, 2,2);
    // bilinear 2x2 -> 7x7
    bilinear_bf16<<<cdiv_i(16*1024*7*7,256), 256, 0, stream>>>(
        bufB, bufA, 16*1024, 2,2, 7,7);
    // locally-connected + leaky ReLU
    localconv_wmma<<<dim3(49,16), 32, 0, stream>>>(bufA, lcw, lcb, bufB);
    // final FC
    fc_wmma<<<cdiv_i(310,8), 256, 0, stream>>>(bufB, fcw, fcb, out);
}
